// Trainer3_52398601012047
// MI455X (gfx1250) — matrix operand order repaired and checked by reading, not compiled, not run on hardware
//
#include <hip/hip_runtime.h>
#include <hip/hip_fp16.h>

typedef __attribute__((ext_vector_type(16))) _Float16 v16h;
typedef __attribute__((ext_vector_type(8)))  _Float16 v8h;
typedef __attribute__((ext_vector_type(4)))  _Float16 v4h;
typedef __attribute__((ext_vector_type(8)))  float    v8f;

// f16 transposed, zero-padded weights in d_ws (offsets in halves)
#define WS_E1 0        // [32][192]
#define WS_E2 6144     // [16][32]
#define WS_D2 6656     // [48][96]
#define WS_D3 11264    // [32][64]  (N pad 24->32, K pad 48->64, zeros)
#define WS_D4 13312    // [32][32]  (N pad 24->32, K pad 24->32, zeros)
#define WS_TOTAL 14336

__device__ __forceinline__ v8f wmma16(v16h a, v16h b, v8f c) {
  return __builtin_amdgcn_wmma_f32_16x16x32_f16(false, a, false, b, (short)0, c,
                                                false, false);
}
__device__ __forceinline__ float sigm(float v) { return 1.0f / (1.0f + __expf(-v)); }

// A operand: lanes 0-15 hold K[0..15] of row M=lane;
// lanes 16-31 hold K[16..31] of row M=lane-16 (matches ldB's contiguous K order).
__device__ __forceinline__ v16h ldA(const _Float16* S, int W, int kc, int l16, int half) {
  const _Float16* row = S + l16 * W + kc * 32;
  return *(const v16h*)(row + half * 16);
}
// B operand: lane holds 16 consecutive K of column n = lane%16 (K base = (lane/16)*16),
// sourced from Bᵀ stored N-major [n][Kpad].
__device__ __forceinline__ v16h ldB(const _Float16* WT, int Kpad, int nt, int kc,
                                    int l16, int half) {
  return *(const v16h*)(WT + (nt * 16 + l16) * Kpad + kc * 32 + half * 16);
}

__global__ void prep_weights(const float* __restrict__ e1_w, const float* __restrict__ e2_w,
                             const float* __restrict__ d2_w, const float* __restrict__ d3_w,
                             const float* __restrict__ d4_w, _Float16* __restrict__ wt) {
  const int t = threadIdx.x;
  for (int i = t; i < WS_TOTAL; i += 256) wt[i] = (_Float16)0.0f;
  __syncthreads();
  for (int i = t; i < 32 * 192; i += 256) { int n = i / 192, k = i % 192;
    wt[WS_E1 + n * 192 + k] = (_Float16)e1_w[k * 32 + n]; }
  for (int i = t; i < 16 * 32; i += 256) { int n = i / 32, k = i % 32;
    wt[WS_E2 + n * 32 + k] = (_Float16)e2_w[k * 16 + n]; }
  for (int i = t; i < 48 * 96; i += 256) { int n = i / 96, k = i % 96;
    wt[WS_D2 + n * 96 + k] = (_Float16)d2_w[k * 48 + n]; }
  for (int i = t; i < 24 * 48; i += 256) { int n = i / 48, k = i % 48;
    wt[WS_D3 + n * 64 + k] = (_Float16)d3_w[k * 24 + n]; }
  for (int i = t; i < 24 * 24; i += 256) { int n = i / 24, k = i % 24;
    wt[WS_D4 + n * 32 + k] = (_Float16)d4_w[k * 24 + n]; }
}

// 128 threads = 4 waves; each wave processes 16 batch elements.
// Per-wave LDS: S (3072 halves) + S2 (1536 halves) = 9216 B; 4 waves = 36864 B.
__global__ __launch_bounds__(128) void gnn_fused(
    const float* __restrict__ x, const float* __restrict__ adj,
    const float* __restrict__ gc1_w, const float* __restrict__ gc1_b,
    const float* __restrict__ gc2_w, const float* __restrict__ gc2_b,
    const float* __restrict__ e1_b, const float* __restrict__ e2_b,
    const float* __restrict__ e3_w, const float* __restrict__ e3_b,
    const float* __restrict__ d1_w, const float* __restrict__ d1_b,
    const float* __restrict__ d2_b, const float* __restrict__ d3_b,
    const float* __restrict__ d4_b,
    const _Float16* __restrict__ wt, float* __restrict__ out)
{
  __shared__ __align__(32) _Float16 smem[4 * 4608];
  const int tid  = threadIdx.x;
  const int wave = tid >> 5;
  const int lane = tid & 31;
  const int l16  = lane & 15;
  const int half = lane >> 4;
  _Float16* S  = smem + wave * 4608;  // 3072 halves: h2 tile [16][192], later reuse
  _Float16* S2 = S + 3072;            // 1536 halves: h1 / e1-out / d1-out / d3-out
  const int tile = blockIdx.x * 4 + wave;
  const int base = tile * 16;

  // ---------------- Phase A: GCN (VALU). 2 lanes per element ----------------
  {
    const int e = base + l16;
    const float* Arow = adj + (size_t)e * 576;
    float xr[24];
    #pragma unroll
    for (int j = 0; j < 24; ++j) xr[j] = x[e * 24 + j];
    float g1w[4], g1b[4];
    #pragma unroll
    for (int k = 0; k < 4; ++k) { g1w[k] = gc1_w[k]; g1b[k] = gc1_b[k]; }
    _Float16* h1 = S2;  // [16][24*4] f16
    for (int i = half * 12; i < half * 12 + 12; ++i) {
      float s = 0.f;
      #pragma unroll
      for (int j = 0; j < 24; ++j) s = fmaf(Arow[i * 24 + j], xr[j], s);
      #pragma unroll
      for (int k = 0; k < 4; ++k)
        h1[l16 * 96 + i * 4 + k] = (_Float16)fmaxf(fmaf(s, g1w[k], g1b[k]), 0.f);
    }
    __syncthreads();
    float g2w[32], g2b[8];
    #pragma unroll
    for (int q = 0; q < 32; ++q) g2w[q] = gc2_w[q];
    #pragma unroll
    for (int f = 0; f < 8; ++f) g2b[f] = gc2_b[f];
    for (int i = half * 12; i < half * 12 + 12; ++i) {   // second adj pass: L2-hot
      float t0 = 0.f, t1 = 0.f, t2 = 0.f, t3 = 0.f;
      #pragma unroll
      for (int j = 0; j < 24; ++j) {
        float a = Arow[i * 24 + j];
        v4h hv = *(const v4h*)(h1 + l16 * 96 + j * 4);
        t0 = fmaf(a, (float)hv[0], t0);
        t1 = fmaf(a, (float)hv[1], t1);
        t2 = fmaf(a, (float)hv[2], t2);
        t3 = fmaf(a, (float)hv[3], t3);
      }
      #pragma unroll
      for (int f = 0; f < 8; ++f) {
        float v = g2b[f];
        v = fmaf(t0, g2w[f], v);      v = fmaf(t1, g2w[8 + f], v);
        v = fmaf(t2, g2w[16 + f], v); v = fmaf(t3, g2w[24 + f], v);
        S[l16 * 192 + i * 8 + f] = (_Float16)fmaxf(v, 0.f);  // h2 flat [16][192]
      }
    }
  }
  __syncthreads();

  // ---------------- e1: [16,192] @ [192,32], relu -> S2 [16][32] f16 ----------------
  {
    v8f c0 = {}, c1 = {};
    #pragma unroll
    for (int kc = 0; kc < 6; ++kc) {
      v16h a = ldA(S, 192, kc, l16, half);
      c0 = wmma16(a, ldB(wt + WS_E1, 192, 0, kc, l16, half), c0);
      c1 = wmma16(a, ldB(wt + WS_E1, 192, 1, kc, l16, half), c1);
    }
    float b0 = e1_b[l16], b1 = e1_b[16 + l16];
    #pragma unroll
    for (int r = 0; r < 8; ++r) {
      int m = r + half * 8;
      S2[m * 32 + l16]      = (_Float16)fmaxf(c0[r] + b0, 0.f);
      S2[m * 32 + 16 + l16] = (_Float16)fmaxf(c1[r] + b1, 0.f);
    }
  }
  __syncthreads();

  // ---------------- e2: [16,32] @ [32,16], relu -> S as f32 [16][16] ----------------
  {
    v8f c = {};
    v16h a = ldA(S2, 32, 0, l16, half);
    c = wmma16(a, ldB(wt + WS_E2, 32, 0, 0, l16, half), c);
    float* Hf = (float*)S;
    float b = e2_b[l16];
    #pragma unroll
    for (int r = 0; r < 8; ++r) {
      int m = r + half * 8;
      Hf[m * 16 + l16] = fmaxf(c[r] + b, 0.f);
    }
  }
  __syncthreads();

  // ------- e3 (K=16,N=5) + sigmoid + d1 (K=5,N=96) + relu: VALU -> S2 [16][96] f16 -------
  {
    const float* Hf = (const float*)S;
    float enc[5];
    #pragma unroll
    for (int o = 0; o < 5; ++o) {
      float s = e3_b[o];
      #pragma unroll
      for (int k = 0; k < 16; ++k) s = fmaf(Hf[l16 * 16 + k], e3_w[k * 5 + o], s);
      enc[o] = sigm(s);
    }
    for (int nn = 0; nn < 48; ++nn) {
      int n = half * 48 + nn;
      float s = d1_b[n];
      #pragma unroll
      for (int o = 0; o < 5; ++o) s = fmaf(enc[o], d1_w[o * 96 + n], s);
      S2[l16 * 96 + n] = (_Float16)fmaxf(s, 0.f);
    }
  }
  __syncthreads();

  // ---------------- d2: [16,96] @ [96,48], relu -> S [16][64] (cols 48..63 = 0) ----------------
  {
    v8f c0 = {}, c1 = {}, c2 = {};
    #pragma unroll
    for (int kc = 0; kc < 3; ++kc) {
      v16h a = ldA(S2, 96, kc, l16, half);
      c0 = wmma16(a, ldB(wt + WS_D2, 96, 0, kc, l16, half), c0);
      c1 = wmma16(a, ldB(wt + WS_D2, 96, 1, kc, l16, half), c1);
      c2 = wmma16(a, ldB(wt + WS_D2, 96, 2, kc, l16, half), c2);
    }
    float b0 = d2_b[l16], b1 = d2_b[16 + l16], b2 = d2_b[32 + l16];
    #pragma unroll
    for (int r = 0; r < 8; ++r) {
      int m = r + half * 8;
      S[m * 64 + l16]      = (_Float16)fmaxf(c0[r] + b0, 0.f);
      S[m * 64 + 16 + l16] = (_Float16)fmaxf(c1[r] + b1, 0.f);
      S[m * 64 + 32 + l16] = (_Float16)fmaxf(c2[r] + b2, 0.f);
      S[m * 64 + 48 + l16] = (_Float16)0.0f;   // K pad for d3
    }
  }
  __syncthreads();

  // ---------------- d3: [16,64] @ [64,32] (24 real), sigmoid -> S2 [16][32] (cols 24..31 = 0) ----------------
  {
    v8f c0 = {}, c1 = {};
    #pragma unroll
    for (int kc = 0; kc < 2; ++kc) {
      v16h a = ldA(S, 64, kc, l16, half);
      c0 = wmma16(a, ldB(wt + WS_D3, 64, 0, kc, l16, half), c0);
      c1 = wmma16(a, ldB(wt + WS_D3, 64, 1, kc, l16, half), c1);
    }
    float b0 = d3_b[l16];
    float b1 = (l16 < 8) ? d3_b[16 + l16] : 0.f;
    #pragma unroll
    for (int r = 0; r < 8; ++r) {
      int m = r + half * 8;
      S2[m * 32 + l16] = (_Float16)sigm(c0[r] + b0);
      S2[m * 32 + 16 + l16] =
          (l16 < 8) ? (_Float16)sigm(c1[r] + b1) : (_Float16)0.0f;  // K pad for d4
    }
  }
  __syncthreads();

  // ---------------- d4: [16,32] @ [32,32] (24 real), sigmoid -> out f32 ----------------
  {
    v8f c0 = {}, c1 = {};
    v16h a = ldA(S2, 32, 0, l16, half);
    c0 = wmma16(a, ldB(wt + WS_D4, 32, 0, 0, l16, half), c0);
    c1 = wmma16(a, ldB(wt + WS_D4, 32, 1, 0, l16, half), c1);
    float b0 = d4_b[l16];
    float b1 = (l16 < 8) ? d4_b[16 + l16] : 0.f;
    #pragma unroll
    for (int r = 0; r < 8; ++r) {
      int m = r + half * 8;
      float* orow = out + (size_t)(base + m) * 24;
      orow[l16] = sigm(c0[r] + b0);
      if (l16 < 8) orow[16 + l16] = sigm(c1[r] + b1);
    }
  }
}

extern "C" void kernel_launch(void* const* d_in, const int* in_sizes, int n_in,
                              void* d_out, int out_size, void* d_ws, size_t ws_size,
                              hipStream_t stream) {
  const float* x     = (const float*)d_in[0];
  const float* adj   = (const float*)d_in[1];
  // d_in[2] = invadj: unused by the reference -> never read (saves 604 MB traffic)
  const float* gc1_w = (const float*)d_in[3];
  const float* gc1_b = (const float*)d_in[4];
  const float* gc2_w = (const float*)d_in[5];
  const float* gc2_b = (const float*)d_in[6];
  const float* e1_w  = (const float*)d_in[7];
  const float* e1_b  = (const float*)d_in[8];
  const float* e2_w  = (const float*)d_in[9];
  const float* e2_b  = (const float*)d_in[10];
  const float* e3_w  = (const float*)d_in[11];
  const float* e3_b  = (const float*)d_in[12];
  const float* d1_w  = (const float*)d_in[13];
  const float* d1_b  = (const float*)d_in[14];
  const float* d2_w  = (const float*)d_in[15];
  const float* d2_b  = (const float*)d_in[16];
  const float* d3_w  = (const float*)d_in[17];
  const float* d3_b  = (const float*)d_in[18];
  const float* d4_w  = (const float*)d_in[19];
  const float* d4_b  = (const float*)d_in[20];
  _Float16* wt = (_Float16*)d_ws;
  float* out = (float*)d_out;

  prep_weights<<<1, 256, 0, stream>>>(e1_w, e2_w, d2_w, d3_w, d4_w, wt);

  const int B = in_sizes[0] / 24;          // 262144
  const int blocks = B / 64;               // 4 waves/block * 16 elements/wave
  gnn_fused<<<blocks, 128, 0, stream>>>(x, adj, gc1_w, gc1_b, gc2_w, gc2_b,
                                        e1_b, e2_b, e3_w, e3_b, d1_w, d1_b,
                                        d2_b, d3_b, d4_b, wt, out);
}
